// MoE_45853070852658
// MI455X (gfx1250) — matrix operand order repaired and checked by reading, not compiled, not run on hardware
//
#include <hip/hip_runtime.h>
#include <hip/hip_bf16.h>
#include <stdint.h>
#include <math.h>

// ---------------------------------------------------------------------------
// MoE forward for gfx1250 (MI455X).
//  - bf16 WMMA (v_wmma_f32_16x16x32_bf16), f32 accumulate.
//  - Weight tiles staged to LDS by the Tensor Data Mover (tensor_load_to_lds),
//    double-buffered, synchronized with s_wait_tensorcnt.
//  - 8 waves per block share one 16-column weight tile (8x reuse), each wave
//    owns a 16-token slice.
// ---------------------------------------------------------------------------

typedef __bf16 bf16;
typedef __attribute__((ext_vector_type(16))) __bf16 v16bf;
typedef __attribute__((ext_vector_type(8)))  __bf16 v8bf;
typedef __attribute__((ext_vector_type(4)))  __bf16 v4bf;
typedef __attribute__((ext_vector_type(2)))  __bf16 v2bf;
typedef __attribute__((ext_vector_type(8)))  float  v8f;
typedef __attribute__((ext_vector_type(4)))  float  v4f;
typedef __attribute__((ext_vector_type(2)))  float  v2f;
typedef __attribute__((ext_vector_type(4)))  unsigned int v4u;
typedef __attribute__((ext_vector_type(8)))  int v8i;
typedef __attribute__((ext_vector_type(4)))  int v4i;

constexpr int HID  = 2048;   // hidden size
constexpr int IMID = 1408;   // routed expert intermediate
constexpr int NEXP = 8;      // experts
constexpr int NTOK = 1024;   // tokens
constexpr int ISH  = 2816;   // shared expert intermediate
constexpr int BPAD = 40;     // padded k-stride (elems) for bf16 LDS B tiles

// ---------------------------------------------------------------------------
// WMMA + fragment helpers
// ---------------------------------------------------------------------------
__device__ inline v8f wmma_bf16(v16bf a, v16bf b, v8f c) {
  return __builtin_amdgcn_wmma_f32_16x16x32_bf16(
      false, a, false, b, (short)0, c, false, false);
}

// A fragment: 16x32 bf16 tile from global, rowbase -> (row0, 0), ld elems.
// lanes 0-15: M=lane, K runs {0..7,16..23}; lanes 16-31: M=lane-16, K {8..15,24..31}
__device__ inline v16bf load_a_frag(const bf16* rowbase, int ld, int k0, int lane) {
  int m  = lane & 15;
  int kb = (lane >> 4) << 3;              // 0 or 8
  const bf16* p = rowbase + (size_t)m * ld + k0 + kb;
  union { v16bf v; v8bf h[2]; } u;
  u.h[0] = *(const v8bf*)(p);             // K = kb .. kb+7
  u.h[1] = *(const v8bf*)(p + 16);        // K = kb+16 .. kb+23
  return u.v;
}

// B fragment from padded bf16 LDS tile sb[col * BPAD + k].
// lane holds column n = lane&15, element j -> K = 8*(lane>>4)+j (j<8), 16+8*(lane>>4)+(j-8) (j>=8).
__device__ inline v16bf load_b_frag_lds(const bf16* sb, int lane) {
  int n  = lane & 15;
  int kh = (lane >> 4) << 3;              // 0 or 8
  const bf16* p = sb + n * BPAD + kh;
  union { v16bf v; v8bf h[2]; } u;
  u.h[0] = *(const v8bf*)(p);
  u.h[1] = *(const v8bf*)(p + 16);
  return u.v;
}

__device__ inline float silu_f(float g) { return g / (1.f + __expf(-g)); }

__device__ inline unsigned lds_off(const void* p) {
  // flat LDS addresses carry the LDS byte offset in the low 32 bits
  return (unsigned)(uintptr_t)p;
}

// ---------------------------------------------------------------------------
// TDM: stage one 16-row x 32-col f32 tile (tile==tensor window) into LDS.
// D# per CDNA5 ISA (group0: count/lds/global/type,
// group1: data_size=4B, dim0=32, dim1=16, tile 32x16, dim0_stride=ld).
// 6-arg builtin variant (groups 2/3 + aux group zeroed, cpol=0).
// ---------------------------------------------------------------------------
__device__ inline void tdm_load_tile16x32_f32(const float* tile, int ld,
                                              unsigned lds_byte_off) {
  unsigned long long ga = (unsigned long long)(uintptr_t)tile;
  v4u g0 = { 1u,                                          // count=1 (valid D#)
             lds_byte_off,                                // lds_addr
             (unsigned)(ga & 0xffffffffull),              // global_addr lo
             (unsigned)((ga >> 32) & 0x1ffffffull)        // global_addr hi
               | (2u << 30) };                            // type=2 (image)
  unsigned uld = (unsigned)ld;
  v8i g1 = { (int)0x00020000u,                 // data_size=2 -> 4 bytes
             (int)(32u << 16),                 // tensor_dim0 lo16 = 32
             (int)(16u << 16),                 // dim0 hi=0 | tensor_dim1 lo16 = 16
             (int)(32u << 16),                 // dim1 hi=0 | tile_dim0 = 32
             (int)16,                          // tile_dim1 = 16 (tile_dim2 = 0)
             (int)uld,                         // tensor_dim0_stride lo32
             0, 0 };
  v4i z4 = { 0, 0, 0, 0 };
  v8i z8 = { 0, 0, 0, 0, 0, 0, 0, 0 };
  __builtin_amdgcn_tensor_load_to_lds(g0, g1, z4, z4, z8, 0);
}

// Cooperative f32->bf16 tile conversion: 2 matrices, 256 threads, 4 elems each.
__device__ inline void convert_tiles2(const float* sf, bf16* sb, int tid) {
  const int m  = tid >> 7;                 // matrix 0/1 (gate/up)
  const int i0 = (tid & 127) * 4;          // 0..508 within 16x32 tile
  const int r  = i0 >> 5;
  const int k  = i0 & 31;
  v4f v = *(const v4f*)(sf + m * 512 + i0);
  v4bf b;
#pragma unroll
  for (int j = 0; j < 4; ++j) b[j] = (bf16)v[j];
  *(v4bf*)(sb + m * (16 * BPAD) + r * BPAD + k) = b;
}

// Single-matrix variant: 256 threads, 2 elems each.
__device__ inline void convert_tile1(const float* sf, bf16* sb, int tid) {
  const int i0 = tid * 2;
  const int r  = i0 >> 5;
  const int k  = i0 & 31;
  v2f v = *(const v2f*)(sf + i0);
  v2bf b;
  b[0] = (bf16)v[0];
  b[1] = (bf16)v[1];
  *(v2bf*)(sb + r * BPAD + k) = b;
}

// ---------------------------------------------------------------------------
// 1) f32 -> bf16 convert (hidden states)
// ---------------------------------------------------------------------------
__global__ __launch_bounds__(256) void k_cvt_bf16(const float* __restrict__ in,
                                                  unsigned short* __restrict__ out,
                                                  int n4) {
  int i = blockIdx.x * 256 + threadIdx.x;
  if (i >= n4) return;
  v4f v = ((const v4f*)in)[i];
  v4bf b;
#pragma unroll
  for (int j = 0; j < 4; ++j) b[j] = (bf16)v[j];
  ((v4bf*)out)[i] = b;
}

// ---------------------------------------------------------------------------
// 2) Router: logits -> softmax -> top-2 -> normalized dense combine [N, E]
// ---------------------------------------------------------------------------
__global__ __launch_bounds__(256) void k_router(const float* __restrict__ x,
                                                const float* __restrict__ gw,
                                                float* __restrict__ combine) {
  __shared__ float red[NEXP][256];
  const int tid   = threadIdx.x;
  const int token = blockIdx.x;
  const float* xr = x + (size_t)token * HID;

  float acc[NEXP];
#pragma unroll
  for (int e = 0; e < NEXP; ++e) acc[e] = 0.f;
  for (int h = tid; h < HID; h += 256) {
    float xv = xr[h];
#pragma unroll
    for (int e = 0; e < NEXP; ++e) acc[e] = fmaf(xv, gw[e * HID + h], acc[e]);
  }
#pragma unroll
  for (int e = 0; e < NEXP; ++e) red[e][tid] = acc[e];
  __syncthreads();
  for (int off = 128; off > 0; off >>= 1) {
    if (tid < off) {
#pragma unroll
      for (int e = 0; e < NEXP; ++e) red[e][tid] += red[e][tid + off];
    }
    __syncthreads();
  }
  if (tid == 0) {
    float p[NEXP];
    float mx = -1e30f;
#pragma unroll
    for (int e = 0; e < NEXP; ++e) mx = fmaxf(mx, red[e][0]);
    float sum = 0.f;
#pragma unroll
    for (int e = 0; e < NEXP; ++e) { p[e] = __expf(red[e][0] - mx); sum += p[e]; }
    float inv = 1.f / sum;
#pragma unroll
    for (int e = 0; e < NEXP; ++e) p[e] *= inv;
    int i1 = 0;
#pragma unroll
    for (int e = 1; e < NEXP; ++e) if (p[e] > p[i1]) i1 = e;
    int i2 = (i1 == 0) ? 1 : 0;
#pragma unroll
    for (int e = 0; e < NEXP; ++e) if (e != i1 && p[e] > p[i2]) i2 = e;
    float w1 = p[i1], w2 = p[i2];
    float norm = 1.f / (w1 + w2 + 1e-20f);
    float* c = combine + (size_t)token * NEXP;
#pragma unroll
    for (int e = 0; e < NEXP; ++e) c[e] = 0.f;
    c[i1] = w1 * norm;
    c[i2] = w2 * norm;
  }
}

// ---------------------------------------------------------------------------
// 3) Routed gate+up.  Block: 1 expert, 1 column tile (16 i's), 128 tokens.
//    Wave w owns tokens [blockRow0 + 16w, +16).  TDM double-buffers the two
//    16x32 f32 weight tiles into LDS; block converts them to bf16 once.
// ---------------------------------------------------------------------------
__global__ __launch_bounds__(256) void k_gateup_routed(
    const unsigned short* __restrict__ xb_raw,
    const float* __restrict__ Wg, const float* __restrict__ Wu,
    const float* __restrict__ combine, unsigned short* __restrict__ hout_raw) {
  __shared__ __align__(16) float sf[2][2][16 * 32];   // [buf][gate/up]
  __shared__ __align__(16) bf16  sb[2][16 * BPAD];    // [gate/up]
  const bf16* xb = (const bf16*)xb_raw;
  bf16* hout = (bf16*)hout_raw;

  const int tid  = threadIdx.x;
  const int lane = tid & 31;
  const int wave = tid >> 5;
  constexpr int IT = IMID / 16;           // 88
  constexpr int MB = NTOK / 128;          // 8
  const int blk = blockIdx.x;
  const int e   = blk / (IT * MB);
  const int rem = blk % (IT * MB);
  const int it  = rem / MB;
  const int mb  = rem % MB;
  const int col0 = it * 16;
  const int row0 = mb * 128 + wave * 16;

  const float* wg = Wg + (size_t)e * IMID * HID + (size_t)col0 * HID;
  const float* wu = Wu + (size_t)e * IMID * HID + (size_t)col0 * HID;
  const bf16*  xr = xb + (size_t)row0 * HID;

  constexpr int STEPS = HID / 32;         // 64
  if (tid < 32) {
    tdm_load_tile16x32_f32(wg, HID, lds_off(&sf[0][0][0]));
    tdm_load_tile16x32_f32(wu, HID, lds_off(&sf[0][1][0]));
  }
  v8f accg = {};
  v8f accu = {};
  for (int i = 0; i < STEPS; ++i) {
    const int cur = i & 1;
    if (tid < 32) {
      if (i + 1 < STEPS) {
        tdm_load_tile16x32_f32(wg + (i + 1) * 32, HID, lds_off(&sf[1 - cur][0][0]));
        tdm_load_tile16x32_f32(wu + (i + 1) * 32, HID, lds_off(&sf[1 - cur][1][0]));
        __builtin_amdgcn_s_wait_tensorcnt(2);   // current buffer landed
      } else {
        __builtin_amdgcn_s_wait_tensorcnt(0);
      }
    }
    __syncthreads();
    convert_tiles2(&sf[cur][0][0], &sb[0][0], tid);
    __syncthreads();
    v16bf a  = load_a_frag(xr, HID, i * 32, lane);
    v16bf bg = load_b_frag_lds(&sb[0][0], lane);
    v16bf bu = load_b_frag_lds(&sb[1][0], lane);
    accg = wmma_bf16(a, bg, accg);
    accu = wmma_bf16(a, bu, accu);
  }

  const int n  = lane & 15;
  const int mh = (lane >> 4) << 3;
#pragma unroll
  for (int rr = 0; rr < 8; ++rr) {
    const int tok = row0 + mh + rr;
    float s = silu_f(accg[rr]) * accu[rr] * combine[tok * NEXP + e];
    hout[(size_t)tok * (NEXP * IMID) + (size_t)e * IMID + col0 + n] = (bf16)s;
  }
}

// ---------------------------------------------------------------------------
// 4) Shared gate+up (same scheme, no expert dim / no combine).
// ---------------------------------------------------------------------------
__global__ __launch_bounds__(256) void k_gateup_shared(
    const unsigned short* __restrict__ xb_raw,
    const float* __restrict__ Wg, const float* __restrict__ Wu,
    unsigned short* __restrict__ hout_raw) {
  __shared__ __align__(16) float sf[2][2][16 * 32];
  __shared__ __align__(16) bf16  sb[2][16 * BPAD];
  const bf16* xb = (const bf16*)xb_raw;
  bf16* hout = (bf16*)hout_raw;

  const int tid  = threadIdx.x;
  const int lane = tid & 31;
  const int wave = tid >> 5;
  constexpr int MB = NTOK / 128;          // 8
  const int blk = blockIdx.x;
  const int it  = blk / MB;
  const int mb  = blk % MB;
  const int col0 = it * 16;
  const int row0 = mb * 128 + wave * 16;

  const float* wg = Wg + (size_t)col0 * HID;
  const float* wu = Wu + (size_t)col0 * HID;
  const bf16*  xr = xb + (size_t)row0 * HID;

  constexpr int STEPS = HID / 32;
  if (tid < 32) {
    tdm_load_tile16x32_f32(wg, HID, lds_off(&sf[0][0][0]));
    tdm_load_tile16x32_f32(wu, HID, lds_off(&sf[0][1][0]));
  }
  v8f accg = {};
  v8f accu = {};
  for (int i = 0; i < STEPS; ++i) {
    const int cur = i & 1;
    if (tid < 32) {
      if (i + 1 < STEPS) {
        tdm_load_tile16x32_f32(wg + (i + 1) * 32, HID, lds_off(&sf[1 - cur][0][0]));
        tdm_load_tile16x32_f32(wu + (i + 1) * 32, HID, lds_off(&sf[1 - cur][1][0]));
        __builtin_amdgcn_s_wait_tensorcnt(2);
      } else {
        __builtin_amdgcn_s_wait_tensorcnt(0);
      }
    }
    __syncthreads();
    convert_tiles2(&sf[cur][0][0], &sb[0][0], tid);
    __syncthreads();
    v16bf a  = load_a_frag(xr, HID, i * 32, lane);
    v16bf bg = load_b_frag_lds(&sb[0][0], lane);
    v16bf bu = load_b_frag_lds(&sb[1][0], lane);
    accg = wmma_bf16(a, bg, accg);
    accu = wmma_bf16(a, bu, accu);
  }

  const int n  = lane & 15;
  const int mh = (lane >> 4) << 3;
#pragma unroll
  for (int rr = 0; rr < 8; ++rr) {
    const int tok = row0 + mh + rr;
    float s = silu_f(accg[rr]) * accu[rr];
    hout[(size_t)tok * ISH + col0 + n] = (bf16)s;
  }
}

// ---------------------------------------------------------------------------
// 5) Down-projection: out = h_sh@Wsd^T + sum_e h_rt[:,e,:]@Wed_e^T
//    (combine folded into h_rt).  9 K-segments, TDM-staged weight tiles.
// ---------------------------------------------------------------------------
__global__ __launch_bounds__(256) void k_down(
    const unsigned short* __restrict__ hsh_raw,
    const unsigned short* __restrict__ hrt_raw,
    const float* __restrict__ Wsd, const float* __restrict__ Wed,
    float* __restrict__ out) {
  __shared__ __align__(16) float sf[2][16 * 32];
  __shared__ __align__(16) bf16  sb[16 * BPAD];
  const bf16* h_sh = (const bf16*)hsh_raw;
  const bf16* h_rt = (const bf16*)hrt_raw;

  const int tid  = threadIdx.x;
  const int lane = tid & 31;
  const int wave = tid >> 5;
  constexpr int MB = NTOK / 128;          // 8
  const int blk = blockIdx.x;
  const int ht  = blk / MB;
  const int mb  = blk % MB;
  const int col0 = ht * 16;
  const int row0 = mb * 128 + wave * 16;

  v8f acc = {};
  for (int seg = 0; seg < 1 + NEXP; ++seg) {
    const bf16* ar;
    int ald;
    const float* wb;
    int wld, steps;
    if (seg == 0) {
      ar = h_sh + (size_t)row0 * ISH;               ald = ISH;
      wb = Wsd + (size_t)col0 * ISH;                wld = ISH;
      steps = ISH / 32;                             // 88
    } else {
      const int e = seg - 1;
      ar = h_rt + (size_t)row0 * (NEXP * IMID) + (size_t)e * IMID;
      ald = NEXP * IMID;
      wb = Wed + (size_t)e * HID * IMID + (size_t)col0 * IMID;
      wld = IMID;
      steps = IMID / 32;                            // 44
    }
    if (tid < 32) tdm_load_tile16x32_f32(wb, wld, lds_off(&sf[0][0]));
    for (int i = 0; i < steps; ++i) {
      const int cur = i & 1;
      if (tid < 32) {
        if (i + 1 < steps) {
          tdm_load_tile16x32_f32(wb + (size_t)(i + 1) * 32, wld,
                                 lds_off(&sf[1 - cur][0]));
          __builtin_amdgcn_s_wait_tensorcnt(1);
        } else {
          __builtin_amdgcn_s_wait_tensorcnt(0);
        }
      }
      __syncthreads();
      convert_tile1(&sf[cur][0], &sb[0], tid);
      __syncthreads();
      v16bf a  = load_a_frag(ar, ald, i * 32, lane);
      v16bf bw = load_b_frag_lds(&sb[0], lane);
      acc = wmma_bf16(a, bw, acc);
    }
  }

  const int n  = lane & 15;
  const int mh = (lane >> 4) << 3;
#pragma unroll
  for (int rr = 0; rr < 8; ++rr) {
    const int tok = row0 + mh + rr;
    out[(size_t)tok * HID + col0 + n] = acc[rr];
  }
}

// ---------------------------------------------------------------------------
// Host launcher
// ---------------------------------------------------------------------------
extern "C" void kernel_launch(void* const* d_in, const int* in_sizes, int n_in,
                              void* d_out, int out_size, void* d_ws, size_t ws_size,
                              hipStream_t stream) {
  const float* x   = (const float*)d_in[0];   // [1,1024,2048]
  const float* gw  = (const float*)d_in[1];   // [8,2048]
  const float* weg = (const float*)d_in[2];   // [8,1408,2048]
  const float* weu = (const float*)d_in[3];   // [8,1408,2048]
  const float* wed = (const float*)d_in[4];   // [8,2048,1408]
  const float* wsg = (const float*)d_in[5];   // [2816,2048]
  const float* wsu = (const float*)d_in[6];   // [2816,2048]
  const float* wsd = (const float*)d_in[7];   // [2048,2816]
  float* out = (float*)d_out;

  char* ws = (char*)d_ws;
  unsigned short* xb = (unsigned short*)ws;                   // [N,H] bf16
  size_t off = (size_t)NTOK * HID * 2;
  float* combine = (float*)(ws + off);                        // [N,E] f32
  off += (size_t)NTOK * NEXP * 4;
  unsigned short* h_rt = (unsigned short*)(ws + off);         // [N,E,I] bf16
  off += (size_t)NTOK * NEXP * IMID * 2;
  unsigned short* h_sh = (unsigned short*)(ws + off);         // [N,I_SH] bf16

  {
    int n4 = NTOK * HID / 4;
    k_cvt_bf16<<<(n4 + 255) / 256, 256, 0, stream>>>(x, xb, n4);
  }
  k_router<<<NTOK, 256, 0, stream>>>(x, gw, combine);
  {
    int blocks = NEXP * (IMID / 16) * (NTOK / 128);           // 5632
    k_gateup_routed<<<blocks, 256, 0, stream>>>(xb, weg, weu, combine, h_rt);
  }
  {
    int blocks = (ISH / 16) * (NTOK / 128);                   // 1408
    k_gateup_shared<<<blocks, 256, 0, stream>>>(xb, wsg, wsu, h_sh);
  }
  {
    int blocks = (HID / 16) * (NTOK / 128);                   // 1024
    k_down<<<blocks, 256, 0, stream>>>(h_sh, h_rt, wsd, wed, out);
  }
}
